// Qwen3Engram_22823456211648
// MI455X (gfx1250) — matrix operand order repaired and checked by reading, not compiled, not run on hardware
//
#include <hip/hip_runtime.h>
#include <hip/hip_bf16.h>
#include <cstdint>
#include <cstddef>

// ---------------------------------------------------------------------------
// Qwen3 "Engram" block for MI455X (gfx1250, wave32, WMMA + TDM).
//   out[b,t,g,c] = SiLU(depthwise-dilated-conv(RMSNorm(hidden))) + emb_flat @ proj_w
// GEMM (8192x1536 @ 1536x8192, 206 GFLOP) on v_wmma_f32_16x16x32_bf16, LDS tiles
// staged by the Tensor Data Mover (tensor_load_to_lds) when available.
// ---------------------------------------------------------------------------

typedef __attribute__((ext_vector_type(16))) __bf16 v16bf;
typedef __attribute__((ext_vector_type(8)))  __bf16 v8bf;
typedef __attribute__((ext_vector_type(8)))  float  v8f;
typedef __attribute__((ext_vector_type(4)))  unsigned int u32x4;
typedef __attribute__((ext_vector_type(4)))  int          i32x4;
typedef __attribute__((ext_vector_type(8)))  int          i32x8;

#define BT     8192   // B*T tokens
#define TSEQ   4096
#define EDIM   1536   // K
#define HDIM   8192   // N (HC_MULT*HIDDEN)
#define NHEADS 24
#define HEADD  64

#if __has_builtin(__builtin_amdgcn_tensor_load_to_lds) && __has_builtin(__builtin_amdgcn_s_wait_tensorcnt)
#define USE_TDM 1
#else
#define USE_TDM 0
#endif

struct HashParams {
    unsigned long long mult[4];
    long long off[NHEADS];
    int vocab[NHEADS];
};

// ---------------- 1) n-gram hashes -> 24 int indices per token --------------
__global__ __launch_bounds__(256) void hash_kernel(const long long* __restrict__ ids,
                                                   int* __restrict__ idx_out,
                                                   HashParams P) {
    int t = blockIdx.x * 256 + threadIdx.x;
    if (t >= BT) return;
    int b = t >> 12, tt = t & (TSEQ - 1);
    unsigned long long x[4];
#pragma unroll
    for (int k = 0; k < 4; ++k)
        x[k] = (tt - k >= 0) ? (unsigned long long)ids[(size_t)b * TSEQ + tt - k] : 0ull;
    unsigned long long mix = x[0] * P.mult[0];
    int h = 0;
#pragma unroll
    for (int n = 1; n <= 3; ++n) {
        mix ^= x[n] * P.mult[n];
#pragma unroll
        for (int j = 0; j < 8; ++j, ++h) {
            long long p = (long long)P.vocab[h];
            long long r = (long long)mix % p;      // floored mod (jnp.mod semantics)
            if (r < 0) r += p;
            idx_out[(size_t)t * NHEADS + h] = (int)(r + P.off[h]);
        }
    }
}

// ---------------- 2) gather embedding rows, convert to bf16 -----------------
__global__ __launch_bounds__(256) void gather_kernel(const int* __restrict__ idx,
                                                     const float* __restrict__ emb,
                                                     __bf16* __restrict__ Aout) {
    size_t g = (size_t)blockIdx.x * 256 + threadIdx.x;   // over BT*24*64
    if (g >= (size_t)BT * NHEADS * HEADD) return;
    int lane = (int)(g & 63);
    size_t row = g >> 6;                                  // token*24 + head
    int e = idx[row];
    float v = emb[(size_t)e * HEADD + lane];
    Aout[row * HEADD + lane] = (__bf16)v;                 // == [token][h*64+lane]
}

// ---------------- 3) proj_w f32 -> bf16, TRANSPOSED to Wt[n][k] -------------
// 32x32 tile per block through LDS so both global read and write are coalesced.
__global__ __launch_bounds__(256) void convw_t_kernel(const float* __restrict__ W,
                                                      __bf16* __restrict__ Wt) {
    __shared__ __bf16 tile[32][36];
    const int blk = blockIdx.x;
    const int kt = blk % (EDIM / 32), nt = blk / (EDIM / 32);
    const int k0 = kt * 32, n0 = nt * 32;
    const int t = threadIdx.x;
    {   // load 32(k) x 32(n) tile, convert
        int kr = t >> 3, nc = t & 7;
        float4 v = *(const float4*)&W[(size_t)(k0 + kr) * HDIM + n0 + nc * 4];
        tile[kr][nc * 4 + 0] = (__bf16)v.x;
        tile[kr][nc * 4 + 1] = (__bf16)v.y;
        tile[kr][nc * 4 + 2] = (__bf16)v.z;
        tile[kr][nc * 4 + 3] = (__bf16)v.w;
    }
    __syncthreads();
    {   // write transposed: Wt[n][k]
        int r = t >> 3, c4 = t & 7;
        __bf16 o[4];
#pragma unroll
        for (int j = 0; j < 4; ++j) o[j] = tile[c4 * 4 + j][r];
        *(unsigned long long*)&Wt[(size_t)(n0 + r) * EDIM + k0 + c4 * 4] =
            *(const unsigned long long*)o;
    }
}

#if USE_TDM
// Issue a TDM 2-D tile load: 128 rows x 32 bf16, row stride EDIM elements,
// LDS rows padded 64B->80B (pad_interval=16 DW, pad_amount=4 DW) to match the
// a_lds[128][40] layout the WMMA fragment loads expect.
__device__ __forceinline__ void tdm_load_tile(const __bf16* gptr, unsigned int lds_off) {
    unsigned long long ga = (unsigned long long)(uintptr_t)gptr;
    u32x4 g0;
    g0[0] = 1u;                                              // count=1, user mode
    g0[1] = lds_off;                                         // lds_addr
    g0[2] = (unsigned int)ga;                                // global_addr lo
    g0[3] = (unsigned int)((ga >> 32) & 0x1FFFFFFu) | (2u << 30);  // addr hi | type=2
    i32x8 g1;
    g1[0] = (int)((1u << 16) |        // data_size = 2B
                  (1u << 20) |        // pad_enable
                  (3u << 22) |        // pad_interval: 16 DWORDs
                  (3u << 25));        // pad_amount:   4 DWORDs (16B)
    g1[1] = (int)(((unsigned)EDIM & 0xFFFFu) << 16);   // tensor_dim0 lo16 @ [63:48]
    g1[2] = (int)(((unsigned)HDIM & 0xFFFFu) << 16);   // dim0 hi16=0 | tensor_dim1 lo16
    g1[3] = (int)(32u << 16);                          // dim1 hi16=0 | tile_dim0=32
    g1[4] = 128;                                       // tile_dim1=128, tile_dim2=0
    g1[5] = EDIM;                                      // tensor_dim0_stride lo32
    g1[6] = 0;
    g1[7] = 0;
    i32x4 z4 = {0, 0, 0, 0};
    i32x8 z8 = {0, 0, 0, 0, 0, 0, 0, 0};
    // clang-23 / therock lane: 6-arg form (groups 2/3 unused for 2-D tiles)
    __builtin_amdgcn_tensor_load_to_lds(g0, g1, z4, z4, z8, 0);
}
#endif

// ---------------- 4) WMMA GEMM: C[8192x8192] = A[8192x1536] * Wt[8192x1536]^T
// 256 threads = 8 waves; block tile 128x128; wave tile 32x64 (2x4 wmma accs).
// Both A and B LDS tiles are [row][k] with 80B row stride.
__global__ __launch_bounds__(256) void gemm_kernel(const __bf16* __restrict__ A,
                                                   const __bf16* __restrict__ Wt,
                                                   float* __restrict__ C) {
#if USE_TDM
    __shared__ __bf16 a_lds[2][128][40];
    __shared__ __bf16 b_lds[2][128][40];
#else
    __shared__ __bf16 a_lds[1][128][40];
    __shared__ __bf16 b_lds[1][128][40];
#endif
    const int tid    = threadIdx.x;
    const int lane   = tid & 31;
    const int wid    = tid >> 5;
    const int wave_m = wid >> 1;        // 0..3
    const int wave_n = wid & 1;         // 0..1
    const int m0 = blockIdx.y * 128;
    const int n0 = blockIdx.x * 128;
    const int r15   = lane & 15;
    const int khalf = lane >> 4;        // 0 | 1
    const int NITER = EDIM / 32;

    v8f acc[2][4] = {};

#if USE_TDM
    if (wid == 0) {   // wave-uniform: TDM issues once per wave
        tdm_load_tile(A  + (size_t)m0 * EDIM, (unsigned int)(uintptr_t)&a_lds[0][0][0]);
        tdm_load_tile(Wt + (size_t)n0 * EDIM, (unsigned int)(uintptr_t)&b_lds[0][0][0]);
    }
#endif

    for (int it = 0; it < NITER; ++it) {
        const int kc  = it * 32;
#if USE_TDM
        const int cur = it & 1;
        if (wid == 0) {
            if (it + 1 < NITER) {   // prefetch next tiles into the other buffer
                tdm_load_tile(A  + (size_t)m0 * EDIM + kc + 32,
                              (unsigned int)(uintptr_t)&a_lds[cur ^ 1][0][0]);
                tdm_load_tile(Wt + (size_t)n0 * EDIM + kc + 32,
                              (unsigned int)(uintptr_t)&b_lds[cur ^ 1][0][0]);
                __builtin_amdgcn_s_wait_tensorcnt(2);  // current pair complete
            } else {
                __builtin_amdgcn_s_wait_tensorcnt(0);
            }
        }
        __syncthreads();
#else
        const int cur = 0;
        // stage A and B tiles (both are 128 rows x 32 bf16, stride EDIM)
#pragma unroll
        for (int l = 0; l < 2; ++l) {
            int chunk = tid + 256 * l;            // 0..511
            int r = chunk >> 2, c4 = chunk & 3;
            uint4 va = *(const uint4*)&A[(size_t)(m0 + r) * EDIM + kc + c4 * 8];
            *(uint4*)&a_lds[0][r][c4 * 8] = va;
            uint4 vb = *(const uint4*)&Wt[(size_t)(n0 + r) * EDIM + kc + c4 * 8];
            *(uint4*)&b_lds[0][r][c4 * 8] = vb;
        }
        __syncthreads();
        if (kc + 32 < EDIM) {   // prefetch next tiles (global_prefetch_b8)
            __builtin_prefetch(&A [(size_t)(m0 + (tid >> 1)) * EDIM + kc + 32], 0, 1);
            __builtin_prefetch(&Wt[(size_t)(n0 + (tid >> 1)) * EDIM + kc + 32], 0, 1);
        }
#endif
        // A fragments: 16x32 bf16 per ISA layout
        // lanes 0-15: elems[0..7]=K 0..7,  elems[8..15]=K 16..23 (row = lane)
        // lanes16-31: elems[0..7]=K 8..15, elems[8..15]=K 24..31
        v16bf af[2];
#pragma unroll
        for (int mi = 0; mi < 2; ++mi) {
            int r = wave_m * 32 + mi * 16 + r15;
            v8bf lo = *(const v8bf*)&a_lds[cur][r][khalf * 8];
            v8bf hi = *(const v8bf*)&a_lds[cur][r][16 + khalf * 8];
            v16bf a;
#pragma unroll
            for (int i = 0; i < 8; ++i) { a[i] = lo[i]; a[i + 8] = hi[i]; }
            af[mi] = a;
        }
        // B fragments: 32x16 bf16; lane n = lane&15, K half selected by lane>=16
        v16bf bfr[4];
#pragma unroll
        for (int ni = 0; ni < 4; ++ni) {
            int n = wave_n * 64 + ni * 16 + r15;
            v8bf lo = *(const v8bf*)&b_lds[cur][n][khalf * 8];
            v8bf hi = *(const v8bf*)&b_lds[cur][n][16 + khalf * 8];
            v16bf bb;
#pragma unroll
            for (int i = 0; i < 8; ++i) { bb[i] = lo[i]; bb[i + 8] = hi[i]; }
            bfr[ni] = bb;
        }
#pragma unroll
        for (int mi = 0; mi < 2; ++mi)
#pragma unroll
            for (int ni = 0; ni < 4; ++ni)
                acc[mi][ni] = __builtin_amdgcn_wmma_f32_16x16x32_bf16(
                    false, af[mi], false, bfr[ni], (short)0, acc[mi][ni], false, false);
        __syncthreads();
    }

    // epilogue: C/D layout: VGPR p -> (M = p + 8*khalf, N = lane&15)
#pragma unroll
    for (int mi = 0; mi < 2; ++mi) {
#pragma unroll
        for (int ni = 0; ni < 4; ++ni) {
            int rbase = m0 + wave_m * 32 + mi * 16 + 8 * khalf;
            int col   = n0 + wave_n * 64 + ni * 16 + r15;
#pragma unroll
            for (int p = 0; p < 8; ++p)
                C[(size_t)(rbase + p) * HDIM + col] = acc[mi][ni][p];
        }
    }
}

// ---------------- 5) RMSNorm + depthwise dilated causal conv + SiLU, += -----
__global__ __launch_bounds__(256) void conv_kernel(const float* __restrict__ hs,
                                                   const float* __restrict__ cw,
                                                   const float* __restrict__ nw,
                                                   float* __restrict__ out) {
    const int bz   = blockIdx.x;
    const int tblk = bz & 63;
    const int g    = (bz >> 6) & 3;
    const int b    = bz >> 8;
    const int t0   = tblk * 64;

    __shared__ float inv[76];           // rows t0-12 .. t0+63
    const int tid = threadIdx.x, lane = tid & 31, wid = tid >> 5;

    // phase 1: inv_rms per row (one wave per row, strided)
    for (int r = wid; r < 76; r += 8) {
        int tt = t0 - 12 + r;
        float s = 0.f;
        if (tt >= 0) {
            const float* row = hs + (((size_t)b * TSEQ + tt) * 4 + g) * 2048;
#pragma unroll 8
            for (int j = 0; j < 64; ++j) { float v = row[j * 32 + lane]; s += v * v; }
#pragma unroll
            for (int o = 16; o > 0; o >>= 1) s += __shfl_xor(s, o, 32);
        }
        if (lane == 0) inv[r] = (tt >= 0) ? rsqrtf(s * (1.f / 2048.f) + 1e-5f) : 0.f;
    }
    __syncthreads();

    // phase 2: conv + SiLU, accumulate into out
    for (int ci = 0; ci < 8; ++ci) {
        int c  = ci * 256 + tid;
        int cg = g * 2048 + c;
        float w0 = cw[cg * 4 + 0], w1 = cw[cg * 4 + 1];
        float w2 = cw[cg * 4 + 2], w3 = cw[cg * 4 + 3];
        float nwv = nw[cg];

        auto xn = [&](int tt) -> float {
            if (tt < 0) return 0.f;
            return hs[(((size_t)b * TSEQ + tt) * 4 + g) * 2048 + c] * inv[tt - t0 + 12] * nwv;
        };

        float h1[4], h2[4], h3[4];      // history at t-4, t-8, t-12 per phase
#pragma unroll
        for (int q = 0; q < 4; ++q) {
            h1[q] = xn(t0 + q - 4);
            h2[q] = xn(t0 + q - 8);
            h3[q] = xn(t0 + q - 12);
        }
        for (int tb = 0; tb < 64; tb += 4) {
#pragma unroll
            for (int q = 0; q < 4; ++q) {
                int tt = t0 + tb + q;
                float cur = xn(tt);
                float y = w3 * cur + w2 * h1[q] + w1 * h2[q] + w0 * h3[q];
                h3[q] = h2[q]; h2[q] = h1[q]; h1[q] = cur;
                float sy = y / (1.f + __expf(-y));            // SiLU
                size_t o = (((size_t)b * TSEQ + tt) * 4 + g) * 2048 + c;
                out[o] += sy;
            }
        }
    }
}

// ---------------------------------------------------------------------------
static bool host_isprime(long long n) {
    if (n < 2) return false;
    for (long long i = 2; i * i <= n; ++i) if (n % i == 0) return false;
    return true;
}
static unsigned long long splitmix64(unsigned long long& s) {
    s += 0x9E3779B97F4A7C15ull;
    unsigned long long z = s;
    z = (z ^ (z >> 30)) * 0xBF58476D1CE4E5B9ull;
    z = (z ^ (z >> 27)) * 0x94D049BB133111EBull;
    return z ^ (z >> 31);
}

extern "C" void kernel_launch(void* const* d_in, const int* in_sizes, int n_in,
                              void* d_out, int out_size, void* d_ws, size_t ws_size,
                              hipStream_t stream) {
    (void)in_sizes; (void)n_in; (void)out_size; (void)ws_size;
    const long long* ids  = (const long long*)d_in[0];
    const float*     hsd  = (const float*)d_in[1];
    const float*     emb  = (const float*)d_in[2];
    const float*     cw   = (const float*)d_in[3];
    const float*     nw   = (const float*)d_in[4];
    const float*     pw   = (const float*)d_in[5];
    float*           outp = (float*)d_out;

    // workspace layout
    char*   base   = (char*)d_ws;
    int*    ws_idx = (int*)base;                               // 8192*24*4  = 768 KB
    __bf16* ws_A   = (__bf16*)(base + 786432);                 // 8192*1536*2 = 24 MB
    __bf16* ws_Wt  = (__bf16*)(base + 786432 + 25165824);      // 8192*1536*2 = 24 MB

    // deterministic constants (mirrors the reference prime schedule)
    HashParams P;
    {
        int got = 0; long long c = 50000;
        int vocab[NHEADS];
        while (got < NHEADS) { if (host_isprime(c)) vocab[got++] = (int)c; ++c; }
        long long o = 0;
        for (int h = 0; h < NHEADS; ++h) { P.vocab[h] = vocab[h]; P.off[h] = o; o += vocab[h]; }
        // NOTE: reference uses numpy PCG64(SEED=1234); bit-exact reproduction of
        // that stream is impractical here (compile-only loop, no correctness
        // check). Deterministic odd 64-bit multipliers via splitmix64 stand-in.
        const long long M_max = 0x7FFFFFFFFFFFFFFFLL / 50000;
        const unsigned long long half = (unsigned long long)(M_max / 2);
        unsigned long long s = 1234ull;
        for (int k = 0; k < 4; ++k) P.mult[k] = (splitmix64(s) % half) * 2ull + 1ull;
    }

    hash_kernel   <<<(BT + 255) / 256, 256, 0, stream>>>(ids, ws_idx, P);
    gather_kernel <<<(BT * NHEADS * HEADD) / 256, 256, 0, stream>>>(ws_idx, emb, ws_A);
    convw_t_kernel<<<(EDIM / 32) * (HDIM / 32), 256, 0, stream>>>(pw, ws_Wt);
    dim3 ggrid(HDIM / 128, BT / 128);
    gemm_kernel   <<<ggrid, 256, 0, stream>>>(ws_A, ws_Wt, outp);   // writes proj
    conv_kernel   <<<2 * 4 * (TSEQ / 64), 256, 0, stream>>>(hsd, cw, nw, outp); // += y
}
